// EncodingBlock_84112639525028
// MI455X (gfx1250) — matrix operand order repaired and checked by reading, not compiled, not run on hardware
//
#include <hip/hip_runtime.h>
#include <hip/hip_bf16.h>

// ---------------------------------------------------------------------------
// EncodingBlock for MI455X (gfx1250): wave32 WMMA f16 (f32 accum) for all
// GEMM-shaped stages, float4 VALU for bandwidth-bound elementwise stages.
// Weights pre-converted to f16 once (L2-resident), staged into LDS as
// uint4 copies; WMMA fragments load as contiguous ds_load_b128 chunks.
// Shapes: N=32, C=64, T=512, V=25, H=3, D=8, BC=16.
// ---------------------------------------------------------------------------

typedef __attribute__((ext_vector_type(16))) _Float16 v16h;
typedef __attribute__((ext_vector_type(8)))  _Float16 v8h;
typedef __attribute__((ext_vector_type(8)))  float    v8f;

#define WMMA16(a, b, c) \
  __builtin_amdgcn_wmma_f32_16x16x32_f16(false, (a), false, (b), (short)0, (c), false, false)

#define KN 32
#define KC 64
#define KT 512
#define KV 25
#define KTV (KT * KV)          // 12800
#define NTV (KN * KTV)         // 409600
#define NBIG ((size_t)KN * KC * KT * KV)  // 26214400 elems
#define EPS 1e-5f

// f16 weight cache layout (element offsets in workspace f16 region)
#define W16_QKW   0           // 64x64 (rows>=48 zero)
#define W16_CONVD 4096        // 3x64x64
#define W16_W1    16384       // 64x64
#define W16_TOTAL 20480

__device__ __forceinline__ v16h cat8(v8h lo, v8h hi) {
  return __builtin_shufflevector(lo, hi, 0, 1, 2, 3, 4, 5, 6, 7,
                                 8, 9, 10, 11, 12, 13, 14, 15);
}

// ---- WMMA fragment loaders (16-bit, wave32 layouts per ISA 7.12.2) --------
// A: 16x32 tile from row-major f16 [M][K]; per lane two contiguous 16B chunks.
__device__ __forceinline__ v16h ldA(const _Float16* S, int ld, int r0, int c0) {
  int lane = threadIdx.x & 31;
  const _Float16* p = S + (r0 + (lane & 15)) * ld + c0 + ((lane & 16) ? 8 : 0);
  return cat8(*(const v8h*)p, *(const v8h*)(p + 16));
}

// A with only K<8 valid (zero-padded to 32).
__device__ __forceinline__ v16h ldA_pad8(const _Float16* S, int ld, int r0, int c0) {
  int lane = threadIdx.x & 31;
  v8h z = {};
  v8h lo = *(const v8h*)(S + (r0 + (lane & 15)) * ld + c0);
  if (lane & 16) lo = z;  // lanes 16-31 cover K=8..15,24..31 -> all padded
  return cat8(lo, z);
}

// B: 32x16 (KxN) tile from N-major f16 [N][K]; per lane two contiguous 16B chunks.
__device__ __forceinline__ v16h ldBnm(const _Float16* S, int ldk, int r0, int c0) {
  int lane = threadIdx.x & 31;
  const _Float16* p = S + (c0 + (lane & 15)) * ldk + r0 + ((lane & 16) ? 8 : 0);
  return cat8(*(const v8h*)p, *(const v8h*)(p + 16));
}

// B with only K<8 valid, read N-major.
__device__ __forceinline__ v16h ldBnm_pad8(const _Float16* S, int ldk, int c0, int k0) {
  int lane = threadIdx.x & 31;
  v8h z = {};
  v8h lo = *(const v8h*)(S + (c0 + (lane & 15)) * ldk + k0);
  if (lane & 16) lo = z;  // lanes 16-31 cover K=8..15,24..31 -> padded
  return cat8(lo, z);
}

// ---------------------------------------------------------------------------
// K0: one-time f32 -> f16 weight conversion into workspace (L2-resident).
// ---------------------------------------------------------------------------
__global__ __launch_bounds__(256)
void k_cvt_weights(const float* __restrict__ qk_w, const float* __restrict__ convd_w,
                   const float* __restrict__ w1, _Float16* __restrict__ wh) {
  for (int idx = blockIdx.x * 256 + threadIdx.x; idx < W16_TOTAL;
       idx += gridDim.x * 256) {
    _Float16 v;
    if (idx < W16_CONVD)
      v = (idx < 48 * 64) ? (_Float16)qk_w[idx] : (_Float16)0.0f;
    else if (idx < W16_W1)
      v = (_Float16)convd_w[idx - W16_CONVD];
    else
      v = (_Float16)w1[idx - W16_W1];
    wh[idx] = v;
  }
}

// ---------------------------------------------------------------------------
// K1: per-(n,t) fused LayerNorm + QK proj + attention + SA-GC (pre-BN).
// One wave32 per block; grid = N*T = 16384.
// ---------------------------------------------------------------------------
__global__ __launch_bounds__(32)
void k_attn_sagc(const float* __restrict__ x, const float* __restrict__ A_topo,
                 const float* __restrict__ ln_g, const float* __restrict__ ln_b,
                 const _Float16* __restrict__ wh, const float* __restrict__ qk_b,
                 const float* __restrict__ convd_b, float* __restrict__ sa_out) {
  const int nt = blockIdx.x;
  const int n = nt / KT, t = nt % KT;
  const int lane = threadIdx.x;

  __shared__ alignas(16) float    sfeat[32][64];    // raw feat, v-major (for LN)
  __shared__ alignas(16) _Float16 sfeatHT[64][32];  // raw feat f16, c-major (B of z-GEMM)
  __shared__ alignas(16) _Float16 syln[32][64];     // layernormed (A of qk-GEMM)
  __shared__ alignas(16) _Float16 sBt[64][64];      // B staging, N-major [out][in]
  __shared__ alignas(16) _Float16 sqk[32][48];      // q|k projections, row-major
  __shared__ alignas(16) float    sdots[32][32];    // attention logits (per head)
  __shared__ alignas(16) _Float16 sA[32][32];       // softmax * topology (per head)
  __shared__ alignas(16) _Float16 sz[32][64];       // z = A @ feat (per head)

  // 1) load feature slice (v innermost -> coalesced 25-elem runs); pad v>=25
  for (int idx = lane; idx < 64 * 32; idx += 32) {
    int c = idx >> 5, v = idx & 31;
    float val = (v < KV) ? x[((n * KC + c) * KT + t) * KV + v] : 0.0f;
    sfeat[v][c] = val;
    sfeatHT[c][v] = (_Float16)val;
  }
  // stage f16 qk_w as N-major B via vector copy
  {
    uint4* dst = (uint4*)&sBt[0][0];
    const uint4* src = (const uint4*)(wh + W16_QKW);
    for (int idx = lane; idx < 512; idx += 32) dst[idx] = src[idx];
  }
  __syncthreads();

  // 2) LayerNorm over C per row (one lane per joint)
  {
    int v = lane;
    if (v < KV) {
      float s = 0.f, ss = 0.f;
      for (int c = 0; c < KC; ++c) { float f = sfeat[v][c]; s += f; ss += f * f; }
      float mu = s * (1.0f / KC);
      float var = ss * (1.0f / KC) - mu * mu;
      float rs = rsqrtf(var + EPS);
      for (int c = 0; c < KC; ++c)
        syln[v][c] = (_Float16)((sfeat[v][c] - mu) * rs * ln_g[c] + ln_b[c]);
    } else {
      for (int c = 0; c < KC; ++c) syln[v][c] = (_Float16)0.0f;
    }
  }
  __syncthreads();

  // 3) qk = y @ qk_w^T + qk_b : 32x48, K=64 -> 12 WMMA
#pragma unroll
  for (int mt = 0; mt < 2; ++mt)
#pragma unroll
    for (int jt = 0; jt < 3; ++jt) {
      v8f acc = {};
#pragma unroll
      for (int kt = 0; kt < 2; ++kt) {
        v16h a = ldA(&syln[0][0], 64, mt * 16, kt * 32);
        v16h b = ldBnm(&sBt[0][0], 64, kt * 32, jt * 16);
        acc = WMMA16(a, b, acc);
      }
      int col = jt * 16 + (lane & 15);
      int rbase = mt * 16 + ((lane & 16) ? 8 : 0);
#pragma unroll
      for (int r = 0; r < 8; ++r)
        sqk[rbase + r][col] = (_Float16)(acc[r] + qk_b[col]);
    }
  __syncthreads();

  v8f acc_sa[2][4];
  {
    v8f z = {};
#pragma unroll
    for (int i = 0; i < 2; ++i)
#pragma unroll
      for (int j = 0; j < 4; ++j) acc_sa[i][j] = z;
  }
  const float scale = 0.35355339059327373f;  // D^-0.5, D=8

  for (int h = 0; h < 3; ++h) {
    // a) dots = q_h @ k_h^T * scale (K=8 zero-padded to 32)
#pragma unroll
    for (int mt = 0; mt < 2; ++mt)
#pragma unroll
      for (int jt = 0; jt < 2; ++jt) {
        v16h a = ldA_pad8(&sqk[0][0], 48, mt * 16, h * 8);
        v16h b = ldBnm_pad8(&sqk[0][0], 48, jt * 16, 24 + h * 8);
        v8f acc = {};
        acc = WMMA16(a, b, acc);
        int col = jt * 16 + (lane & 15);
        int rbase = mt * 16 + ((lane & 16) ? 8 : 0);
#pragma unroll
        for (int r = 0; r < 8; ++r) sdots[rbase + r][col] = acc[r] * scale;
      }
    __syncthreads();

    // b) softmax over valid j, multiplied by topology prior
    {
      int i = lane;
      if (i < KV) {
        float mx = -1e30f;
        for (int j = 0; j < KV; ++j) mx = fmaxf(mx, sdots[i][j]);
        float sum = 0.f;
        for (int j = 0; j < KV; ++j) sum += expf(sdots[i][j] - mx);
        float inv = 1.0f / sum;
        for (int j = 0; j < 32; ++j)
          sA[i][j] = (j < KV)
              ? (_Float16)(expf(sdots[i][j] - mx) * inv * A_topo[(h * KV + i) * KV + j])
              : (_Float16)0.0f;
      } else {
        for (int j = 0; j < 32; ++j) sA[i][j] = (_Float16)0.0f;
      }
    }
    __syncthreads();

    // c) z_h = A_h @ feat : 32x64, K=32 -> 8 WMMA (B = c-major feat)
#pragma unroll
    for (int mt = 0; mt < 2; ++mt)
#pragma unroll
      for (int ct = 0; ct < 4; ++ct) {
        v16h a = ldA(&sA[0][0], 32, mt * 16, 0);
        v16h b = ldBnm(&sfeatHT[0][0], 32, 0, ct * 16);
        v8f acc = {};
        acc = WMMA16(a, b, acc);
        int col = ct * 16 + (lane & 15);
        int rbase = mt * 16 + ((lane & 16) ? 8 : 0);
#pragma unroll
        for (int r = 0; r < 8; ++r) sz[rbase + r][col] = (_Float16)acc[r];
      }
    __syncthreads();

    // d) stage f16 convd_w[h] as N-major B via vector copy
    {
      uint4* dst = (uint4*)&sBt[0][0];
      const uint4* src = (const uint4*)(wh + W16_CONVD + h * 4096);
      for (int idx = lane; idx < 512; idx += 32) dst[idx] = src[idx];
    }
    __syncthreads();

    // e) sa += z_h @ W_h^T : 32x64, K=64 -> 16 WMMA, accumulate across heads
#pragma unroll
    for (int mt = 0; mt < 2; ++mt)
#pragma unroll
      for (int ot = 0; ot < 4; ++ot) {
        v8f acc = acc_sa[mt][ot];
#pragma unroll
        for (int kt = 0; kt < 2; ++kt) {
          v16h a = ldA(&sz[0][0], 64, mt * 16, kt * 32);
          v16h b = ldBnm(&sBt[0][0], 64, kt * 32, ot * 16);
          acc = WMMA16(a, b, acc);
        }
        acc_sa[mt][ot] = acc;
      }
    __syncthreads();
  }

  // epilogue: + sum_h convd_b[h], store (N,C,T,V)
#pragma unroll
  for (int mt = 0; mt < 2; ++mt)
#pragma unroll
    for (int ot = 0; ot < 4; ++ot) {
      int col = ot * 16 + (lane & 15);
      float bsum = convd_b[col] + convd_b[64 + col] + convd_b[128 + col];
      int rbase = mt * 16 + ((lane & 16) ? 8 : 0);
#pragma unroll
      for (int r = 0; r < 8; ++r) {
        int i = rbase + r;
        if (i < KV)
          sa_out[((n * KC + col) * KT + t) * KV + i] = acc_sa[mt][ot][r] + bsum;
      }
    }
}

// ---------------------------------------------------------------------------
// Per-channel batch statistics (mean, biased var) over (N,T,V); float4 reads.
// One block per channel; deterministic tree reduction (no float atomics).
// ---------------------------------------------------------------------------
__global__ __launch_bounds__(256)
void k_stats(const float* __restrict__ buf, int Ctot, int c0,
             float* __restrict__ mean, float* __restrict__ var) {
  int c = c0 + blockIdx.x;
  float s = 0.f, ss = 0.f;
  for (int i4 = threadIdx.x; i4 < NTV / 4; i4 += 256) {
    int nn = i4 / (KTV / 4), rr = i4 % (KTV / 4);
    float4 xv = *(const float4*)&buf[((size_t)nn * Ctot + c) * KTV + rr * 4];
    s += xv.x + xv.y + xv.z + xv.w;
    ss += xv.x * xv.x + xv.y * xv.y + xv.z * xv.z + xv.w * xv.w;
  }
  __shared__ float rs[256], rss[256];
  rs[threadIdx.x] = s; rss[threadIdx.x] = ss;
  __syncthreads();
  for (int off = 128; off > 0; off >>= 1) {
    if (threadIdx.x < off) {
      rs[threadIdx.x] += rs[threadIdx.x + off];
      rss[threadIdx.x] += rss[threadIdx.x + off];
    }
    __syncthreads();
  }
  if (threadIdx.x == 0) {
    float m = rs[0] * (1.0f / NTV);
    mean[blockIdx.x] = m;
    var[blockIdx.x] = rss[0] * (1.0f / NTV) - m * m;
  }
}

// BN + residual + relu : y_gc = relu(BN(sa) + x), float4.
__global__ __launch_bounds__(256)
void k_bn_res_relu(const float* __restrict__ sa, const float* __restrict__ x,
                   const float* __restrict__ mean, const float* __restrict__ var,
                   const float* __restrict__ g, const float* __restrict__ b,
                   float* __restrict__ out) {
  const size_t total4 = NBIG / 4;
  for (size_t g4 = (size_t)blockIdx.x * blockDim.x + threadIdx.x; g4 < total4;
       g4 += (size_t)gridDim.x * blockDim.x) {
    size_t idx = g4 * 4;
    int c = (int)((idx / KTV) & 63);
    float sc = rsqrtf(var[c] + EPS) * g[c];
    float sh = b[c] - mean[c] * sc;
    float4 sv = *(const float4*)&sa[idx];
    float4 xv = *(const float4*)&x[idx];
    float4 o;
    o.x = fmaxf(sv.x * sc + sh + xv.x, 0.f);
    o.y = fmaxf(sv.y * sc + sh + xv.y, 0.f);
    o.z = fmaxf(sv.z * sc + sh + xv.z, 0.f);
    o.w = fmaxf(sv.w * sc + sh + xv.w, 0.f);
    *(float4*)&out[idx] = o;
  }
}

// ---------------------------------------------------------------------------
// K3: pre = einsum('nctv,koc->nkotv') + b1 as a (N*T*V, 64) x (64, 64) GEMM.
// One wave32 per 32 rows; grid = 12800.
// ---------------------------------------------------------------------------
__global__ __launch_bounds__(32)
void k_tcn_pre(const float* __restrict__ ygc, const _Float16* __restrict__ wh,
               const float* __restrict__ b1, float* __restrict__ pre) {
  const int r0 = blockIdx.x * 32;
  const int lane = threadIdx.x;
  __shared__ alignas(16) _Float16 sin_[32][64];
  __shared__ alignas(16) _Float16 sw[64][64];   // N-major: [ko][c] == memory order

  for (int idx = lane; idx < 32 * 64; idx += 32) {
    int rr = idx >> 6, c = idx & 63;
    int r = r0 + rr;
    int nn = r / KTV, rem = r % KTV, t = rem / KV, v = rem % KV;
    sin_[rr][c] = (_Float16)ygc[((nn * KC + c) * KT + t) * KV + v];
  }
  {
    uint4* dst = (uint4*)&sw[0][0];
    const uint4* src = (const uint4*)(wh + W16_W1);
    for (int idx = lane; idx < 512; idx += 32) dst[idx] = src[idx];
  }
  __syncthreads();

#pragma unroll
  for (int mt = 0; mt < 2; ++mt)
#pragma unroll
    for (int ot = 0; ot < 4; ++ot) {
      v8f acc = {};
#pragma unroll
      for (int kt = 0; kt < 2; ++kt) {
        v16h a = ldA(&sin_[0][0], 64, mt * 16, kt * 32);
        v16h b = ldBnm(&sw[0][0], 64, kt * 32, ot * 16);
        acc = WMMA16(a, b, acc);
      }
      int ko = ot * 16 + (lane & 15);
      int rbase = mt * 16 + ((lane & 16) ? 8 : 0);
#pragma unroll
      for (int r = 0; r < 8; ++r) {
        int rr = r0 + rbase + r;
        int nn = rr / KTV, rem = rr % KTV, t = rem / KV, v = rem % KV;
        pre[((nn * KC + ko) * KT + t) * KV + v] = acc[r] + b1[ko];
      }
    }
}

// BN1 + split: branches 0..2 -> relu into act (N,48,T,V); branch 3 -> cat[:,48:64]
__global__ __launch_bounds__(256)
void k_bn1_split(const float* __restrict__ pre, const float* __restrict__ m1,
                 const float* __restrict__ v1, const float* __restrict__ g1,
                 const float* __restrict__ b1, float* __restrict__ act,
                 float* __restrict__ cat) {
  const size_t total4 = NBIG / 4;
  for (size_t g4 = (size_t)blockIdx.x * blockDim.x + threadIdx.x; g4 < total4;
       g4 += (size_t)gridDim.x * blockDim.x) {
    size_t idx = g4 * 4;
    int nn = (int)(idx / ((size_t)KC * KTV));
    int c = (int)((idx / KTV) & 63);
    int r = (int)(idx % KTV);
    float sc = rsqrtf(v1[c] + EPS) * g1[c];
    float sh = b1[c] - m1[c] * sc;
    float4 pv = *(const float4*)&pre[idx];
    float4 val;
    val.x = pv.x * sc + sh; val.y = pv.y * sc + sh;
    val.z = pv.z * sc + sh; val.w = pv.w * sc + sh;
    int k = c >> 4, o = c & 15;
    if (k < 3) {
      float4 rv;
      rv.x = fmaxf(val.x, 0.f); rv.y = fmaxf(val.y, 0.f);
      rv.z = fmaxf(val.z, 0.f); rv.w = fmaxf(val.w, 0.f);
      *(float4*)&act[((size_t)nn * 48 + c) * KTV + r] = rv;
    } else {
      *(float4*)&cat[((size_t)nn * KC + 48 + o) * KTV + r] = val;
    }
  }
}

// Dilated temporal conv (kernel 5, pad 2*d), per-element: 16ch x 5tap MAC
__global__ __launch_bounds__(256)
void k_tconv(const float* __restrict__ act, const float* __restrict__ w2,
             const float* __restrict__ b2, float* __restrict__ cat, int br, int dil) {
  __shared__ float sw[16 * 16 * 5];
  for (int idx = threadIdx.x; idx < 1280; idx += 256) sw[idx] = w2[br * 1280 + idx];
  __syncthreads();
  const size_t total = (size_t)KN * 16 * KTV;
  for (size_t idx = (size_t)blockIdx.x * blockDim.x + threadIdx.x; idx < total;
       idx += (size_t)gridDim.x * blockDim.x) {
    int nn = (int)(idx / ((size_t)16 * KTV));
    int o = (int)((idx / KTV) & 15);
    int r = (int)(idx % KTV);
    int t = r / KV, v = r % KV;
    float acc = b2[br * 16 + o];
    for (int ic = 0; ic < 16; ++ic) {
      const float* in = &act[((size_t)nn * 48 + br * 16 + ic) * KTV + v];
#pragma unroll
      for (int kk = 0; kk < 5; ++kk) {
        int tt = t + (kk - 2) * dil;
        if (tt >= 0 && tt < KT) acc += sw[(o * 16 + ic) * 5 + kk] * in[tt * KV];
      }
    }
    cat[((size_t)nn * KC + br * 16 + o) * KTV + r] = acc;
  }
}

// Max-pool-3 along T on branch 2 activations -> cat[:,32:48]
__global__ __launch_bounds__(256)
void k_maxpool(const float* __restrict__ act, float* __restrict__ cat) {
  const size_t total = (size_t)KN * 16 * KTV;
  for (size_t idx = (size_t)blockIdx.x * blockDim.x + threadIdx.x; idx < total;
       idx += (size_t)gridDim.x * blockDim.x) {
    int nn = (int)(idx / ((size_t)16 * KTV));
    int o = (int)((idx / KTV) & 15);
    int r = (int)(idx % KTV);
    int t = r / KV, v = r % KV;
    const float* in = &act[((size_t)nn * 48 + 32 + o) * KTV + v];
    float m = in[t * KV];
    if (t > 0) m = fmaxf(m, in[(t - 1) * KV]);
    if (t < KT - 1) m = fmaxf(m, in[(t + 1) * KV]);
    cat[((size_t)nn * KC + 32 + o) * KTV + r] = m;
  }
}

// Final: out = relu(BN2(cat[:,0:48]) | cat[:,48:64]  + x), float4.
__global__ __launch_bounds__(256)
void k_final(const float* __restrict__ cat, const float* __restrict__ x,
             const float* __restrict__ m2, const float* __restrict__ v2,
             const float* __restrict__ g2, const float* __restrict__ b2,
             float* __restrict__ out) {
  const size_t total4 = NBIG / 4;
  for (size_t g4 = (size_t)blockIdx.x * blockDim.x + threadIdx.x; g4 < total4;
       g4 += (size_t)gridDim.x * blockDim.x) {
    size_t idx = g4 * 4;
    int c = (int)((idx / KTV) & 63);
    float sc = 1.0f, sh = 0.0f;
    if (c < 48) {
      sc = rsqrtf(v2[c] + EPS) * g2[c];
      sh = b2[c] - m2[c] * sc;
    }
    float4 cv = *(const float4*)&cat[idx];
    float4 xv = *(const float4*)&x[idx];
    float4 o;
    o.x = fmaxf(cv.x * sc + sh + xv.x, 0.f);
    o.y = fmaxf(cv.y * sc + sh + xv.y, 0.f);
    o.z = fmaxf(cv.z * sc + sh + xv.z, 0.f);
    o.w = fmaxf(cv.w * sc + sh + xv.w, 0.f);
    *(float4*)&out[idx] = o;
  }
}

// ---------------------------------------------------------------------------
extern "C" void kernel_launch(void* const* d_in, const int* in_sizes, int n_in,
                              void* d_out, int out_size, void* d_ws, size_t ws_size,
                              hipStream_t stream) {
  const float* x        = (const float*)d_in[0];
  const float* A_topo   = (const float*)d_in[1];
  const float* ln_g     = (const float*)d_in[2];
  const float* ln_b     = (const float*)d_in[3];
  const float* qk_w     = (const float*)d_in[4];
  const float* qk_b     = (const float*)d_in[5];
  const float* convd_w  = (const float*)d_in[6];
  const float* convd_b  = (const float*)d_in[7];
  const float* sagc_g   = (const float*)d_in[8];
  const float* sagc_b   = (const float*)d_in[9];
  const float* tcn_w1   = (const float*)d_in[10];
  const float* tcn_b1   = (const float*)d_in[11];
  const float* bn1_g    = (const float*)d_in[12];
  const float* bn1_b    = (const float*)d_in[13];
  const float* tcn_w2   = (const float*)d_in[14];
  const float* tcn_b2   = (const float*)d_in[15];
  const float* bn2_g    = (const float*)d_in[16];
  const float* bn2_b    = (const float*)d_in[17];
  float* out = (float*)d_out;

  float* ws = (float*)d_ws;
  float* bufA = ws;              // sa, then pre
  float* bufB = ws + NBIG;       // y_gc, then act (48ch)
  float* bufC = ws + 2 * NBIG;   // concat staging (64ch)
  float* st   = ws + 3 * NBIG;   // stats
  float* m0 = st;       float* v0 = st + 64;
  float* m1 = st + 128; float* v1 = st + 192;
  float* m2 = st + 256; float* v2 = st + 320;
  _Float16* wh = (_Float16*)(st + 512);  // f16 weight cache (16B aligned)

  // 0) one-time weight conversion to f16 (stays L2-resident)
  k_cvt_weights<<<20, 256, 0, stream>>>(qk_w, convd_w, tcn_w1, wh);
  // 1) attention + SA-GC (pre-BN) -> bufA
  k_attn_sagc<<<KN * KT, 32, 0, stream>>>(x, A_topo, ln_g, ln_b, wh, qk_b,
                                          convd_b, bufA);
  // 2) BN stats + BN/residual/relu -> bufB (y_gc)
  k_stats<<<64, 256, 0, stream>>>(bufA, 64, 0, m0, v0);
  k_bn_res_relu<<<4096, 256, 0, stream>>>(bufA, x, m0, v0, sagc_g, sagc_b, bufB);
  // 3) channel-mix GEMM (pre) -> bufA
  k_tcn_pre<<<NTV / 32, 32, 0, stream>>>(bufB, wh, tcn_b1, bufA);
  // 4) BN1 stats + split into act (bufB) and plain branch (bufC[:,48:64])
  k_stats<<<64, 256, 0, stream>>>(bufA, 64, 0, m1, v1);
  k_bn1_split<<<4096, 256, 0, stream>>>(bufA, m1, v1, bn1_g, bn1_b, bufB, bufC);
  // 5) temporal branches -> bufC[:,0:48]
  k_tconv<<<4096, 256, 0, stream>>>(bufB, tcn_w2, tcn_b2, bufC, 0, 1);
  k_tconv<<<4096, 256, 0, stream>>>(bufB, tcn_w2, tcn_b2, bufC, 1, 2);
  k_maxpool<<<4096, 256, 0, stream>>>(bufB, bufC);
  // 6) branch BN stats + final residual relu -> out
  k_stats<<<48, 256, 0, stream>>>(bufC, 64, 0, m2, v2);
  k_final<<<4096, 256, 0, stream>>>(bufC, x, m2, v2, bn2_g, bn2_b, out);
}
